// Block_24343874634443
// MI455X (gfx1250) — matrix operand order repaired and checked by reading, not compiled, not run on hardware
//
#include <hip/hip_runtime.h>
#include <hip/hip_bf16.h>
#include <math.h>

#define DIM    768
#define HEADS  12
#define HIDDEN 3072
#define BATCH  8
#define SEQ    1024
#define ROWS   (BATCH*SEQ)   /* 8192 */

typedef __bf16 bf16_t;
typedef __attribute__((ext_vector_type(16))) __bf16 bf16x16;
typedef __attribute__((ext_vector_type(8)))  __bf16 bf16x8;
typedef __attribute__((ext_vector_type(8)))  float  fx8;
typedef unsigned uint4v __attribute__((ext_vector_type(4)));
typedef int      int8v  __attribute__((ext_vector_type(8)));
typedef int      int4v  __attribute__((ext_vector_type(4)));

#if defined(__has_builtin)
#if __has_builtin(__builtin_amdgcn_tensor_load_to_lds) && \
    __has_builtin(__builtin_amdgcn_s_wait_tensorcnt)
#define HAVE_TDM 1
#endif
#endif
#ifndef HAVE_TDM
#define HAVE_TDM 0
#endif

#define USE_ASYNC_LDS 1   /* global_load_async_to_lds_b128 for GEMM A tiles */

static __device__ __forceinline__ bf16_t f2bf(float f) {
  unsigned u = __builtin_bit_cast(unsigned, f);
  unsigned r = (u + 0x7fffu + ((u >> 16) & 1u)) >> 16;   // round-to-nearest-even
  unsigned short s = (unsigned short)r;
  return __builtin_bit_cast(bf16_t, s);
}

static __device__ __forceinline__ bf16x16 mk16(bf16x8 lo, bf16x8 hi) {
  bf16x16 r;
#pragma unroll
  for (int i = 0; i < 8; ++i) { r[i] = lo[i]; r[8 + i] = hi[i]; }
  return r;
}

static __device__ __forceinline__ fx8 wmma_bf16(bf16x16 a, bf16x16 b, fx8 c) {
  return __builtin_amdgcn_wmma_f32_16x16x32_bf16(false, a, false, b, (short)0, c,
                                                 false, false);
}

// Per-lane async copy: 16B global -> 16B LDS, tracked by ASYNCcnt.
// Low 32 bits of a generic pointer to LDS are the LDS byte address
// (flat aperture).
static __device__ __forceinline__ void async_ld_b128(const bf16_t* gp, bf16_t* lp) {
#if USE_ASYNC_LDS
  unsigned lds = (unsigned)(uintptr_t)lp;
  unsigned long long ga = (unsigned long long)(uintptr_t)gp;
  asm volatile("global_load_async_to_lds_b128 %0, %1, off"
               :: "v"(lds), "v"(ga) : "memory");
#else
  *(bf16x8*)lp = *(const bf16x8*)gp;
#endif
}
static __device__ __forceinline__ void wait_async0() {
#if USE_ASYNC_LDS
  asm volatile("s_wait_asynccnt 0x0" ::: "memory");
#endif
}

#if HAVE_TDM
// TDM 2D tile load: [tile1 x tile0] elements of 2B from row-major tensor
// (row length stride0 elems) into LDS, optionally padding pad_amount DWORDs
// after every (2^(padIntCode+1)) DWORDs stored.
static __device__ __forceinline__ void tdm_load_2d(const bf16_t* gsrc, bf16_t* ldst,
                                                   unsigned dim0, unsigned dim1,
                                                   unsigned stride0,
                                                   unsigned tile0, unsigned tile1,
                                                   unsigned padIntCode,
                                                   unsigned padAmtCode,
                                                   unsigned padEnable) {
  unsigned long long ga = (unsigned long long)(uintptr_t)gsrc;
  unsigned lds = (unsigned)(uintptr_t)ldst;
  uint4v g0;
  g0[0] = 1u;                                          // count=1, user mode
  g0[1] = lds;                                         // lds_addr
  g0[2] = (unsigned)(ga & 0xffffffffu);                // global_addr[31:0]
  g0[3] = (unsigned)((ga >> 32) & 0x1ffffffu) | (2u << 30);  // addr[56:32] | type=2
  int8v g1;
  unsigned w0 = (1u << 16)                              // data_size = 2B
              | (padEnable << 20)
              | (padIntCode << 22)
              | (padAmtCode << 25);
  g1[0] = (int)w0;
  g1[1] = (int)((dim0 & 0xffffu) << 16);               // tensor_dim0[15:0] @ bit48
  g1[2] = (int)((dim0 >> 16) | ((dim1 & 0xffffu) << 16));
  g1[3] = (int)((dim1 >> 16) | (tile0 << 16));         // tile_dim0 @ bit112
  g1[4] = (int)(tile1);                                // tile_dim1 (tile_dim2=0)
  g1[5] = (int)stride0;                                // tensor_dim0_stride[31:0]
  g1[6] = 0;                                           // stride0[47:32]=0, stride1 lo=0
  g1[7] = 0;
  int4v z4 = {0, 0, 0, 0};
  int8v z8 = {0, 0, 0, 0, 0, 0, 0, 0};
  __builtin_amdgcn_tensor_load_to_lds(g0, g1, z4, z4, z8, 0);
}
#endif

// ---------------------------------------------------------------------------
// fp32 -> bf16 conversion (weights)
// ---------------------------------------------------------------------------
__global__ void cvt_bf16(const float* __restrict__ src, bf16_t* __restrict__ dst, int n) {
  int i = blockIdx.x * 256 + threadIdx.x;
  if (i < n) dst[i] = f2bf(src[i]);
}

// ---------------------------------------------------------------------------
// LayerNorm: one 256-thread block per row of 768, bf16 output
// ---------------------------------------------------------------------------
__global__ __launch_bounds__(256)
void ln_bf16(const float* __restrict__ x, const float* __restrict__ g,
             const float* __restrict__ b, bf16_t* __restrict__ out) {
  __shared__ float rs1[8], rs2[8];
  const int row = blockIdx.x, tid = threadIdx.x;
  const float* xr = x + (size_t)row * DIM;
  float v[3];
#pragma unroll
  for (int j = 0; j < 3; ++j) v[j] = xr[tid + j * 256];
  float s = v[0] + v[1] + v[2];
  float q = v[0] * v[0] + v[1] * v[1] + v[2] * v[2];
#pragma unroll
  for (int o = 16; o; o >>= 1) { s += __shfl_xor(s, o); q += __shfl_xor(q, o); }
  if ((tid & 31) == 0) { rs1[tid >> 5] = s; rs2[tid >> 5] = q; }
  __syncthreads();
  float S = 0.f, Q = 0.f;
#pragma unroll
  for (int i = 0; i < 8; ++i) { S += rs1[i]; Q += rs2[i]; }
  const float mu   = S * (1.0f / DIM);
  const float var  = Q * (1.0f / DIM) - mu * mu;
  const float rstd = rsqrtf(var + 1e-5f);
  bf16_t* orow = out + (size_t)row * DIM;
#pragma unroll
  for (int j = 0; j < 3; ++j) {
    int c = tid + j * 256;
    orow[c] = f2bf((v[j] - mu) * rstd * g[c] + b[c]);
  }
}

// ---------------------------------------------------------------------------
// bf16 WMMA GEMM: C[M,N] = A[M,K] x B[K,N], fp32 accumulate, fused epilogues.
// 64x256 workgroup tile, 8 waves x (32x64), double-buffered LDS:
//   A tile via global_load_async_to_lds_b128 (ASYNCcnt),
//   B tile register-staged + transposed into LDS during compute.
// ---------------------------------------------------------------------------
#define BM 64
#define BN 256
#define BK 32
#define LDA 40   /* BK + 8 bf16 pad */
#define LDB 40

constexpr int EPI_QKV      = 0;  // scatter into q/k (row-major [n][d]) and vT ([d][n])
constexpr int EPI_BIAS_RES = 1;  // fp32 out = acc + bias + residual
constexpr int EPI_GELU_BF  = 2;  // bf16 out = gelu(acc + bias)

template <int EPI>
__global__ __launch_bounds__(256)
void gemm_bf16_wmma(const bf16_t* __restrict__ A, const bf16_t* __restrict__ Bw,
                    int M, int N, int K,
                    const float* __restrict__ bias, const float* __restrict__ resid,
                    float* __restrict__ outF, bf16_t* __restrict__ outBf,
                    bf16_t* __restrict__ qb, bf16_t* __restrict__ kb,
                    bf16_t* __restrict__ vtb) {
  __shared__ bf16_t As[2][BM * LDA];
  __shared__ bf16_t Bt[2][BN * LDB];   // B tile stored transposed: [n][k]

  const int tid  = threadIdx.x;
  const int m0   = blockIdx.y * BM;
  const int n0   = blockIdx.x * BN;
  const int wid  = tid >> 5, lane = tid & 31;
  const int wm   = wid & 1,  wn   = wid >> 1;
  const int half = lane >> 4, nl  = lane & 15;

  const int am  = tid >> 2;          // 0..63 : A row within tile
  const int ak8 = (tid & 3) * 8;     // 0,8,16,24 : 8-elem chunk within row

  fx8 acc[2][4];
#pragma unroll
  for (int i = 0; i < 2; ++i)
#pragma unroll
    for (int j = 0; j < 4; ++j) acc[i][j] = (fx8)0.0f;

  bf16x8 breg[4];

  // ---- prologue: tile 0
  async_ld_b128(A + (size_t)(m0 + am) * K + ak8, &As[0][am * LDA + ak8]);
#pragma unroll
  for (int j = 0; j < 4; ++j) {
    int c = tid + j * 256, kk = c >> 5, n8 = (c & 31) * 8;
    breg[j] = *(const bf16x8*)&Bw[(size_t)kk * N + n0 + n8];
  }
#pragma unroll
  for (int j = 0; j < 4; ++j) {
    int c = tid + j * 256, kk = c >> 5, n8 = (c & 31) * 8;
#pragma unroll
    for (int e = 0; e < 8; ++e) Bt[0][(n8 + e) * LDB + kk] = breg[j][e];
  }
  wait_async0();
  __syncthreads();

  const int T = K / BK;
  for (int kt = 0; kt < T; ++kt) {
    const int p = kt & 1;
    const bool more = (kt + 1 < T);
    if (more) {
      const int k1 = (kt + 1) * BK;
      async_ld_b128(A + (size_t)(m0 + am) * K + k1 + ak8,
                    &As[p ^ 1][am * LDA + ak8]);
#pragma unroll
      for (int j = 0; j < 4; ++j) {
        int c = tid + j * 256, kk = c >> 5, n8 = (c & 31) * 8;
        breg[j] = *(const bf16x8*)&Bw[(size_t)(k1 + kk) * N + n0 + n8];
      }
      if (kt + 2 < T) {   // stream hint two tiles ahead -> global_prefetch
        __builtin_prefetch(&Bw[(size_t)((kt + 2) * BK) * N + n0 + tid * 8], 0, 1);
        __builtin_prefetch(&A[(size_t)(m0 + am) * K + (kt + 2) * BK + ak8], 0, 1);
      }
    }

    // ---- compute on buffer p
    bf16x16 af[2], bfr[4];
#pragma unroll
    for (int mi = 0; mi < 2; ++mi) {
      const bf16_t* pA = &As[p][(wm * 32 + mi * 16 + nl) * LDA + half * 8];
      af[mi] = mk16(*(const bf16x8*)pA, *(const bf16x8*)(pA + 16));
    }
#pragma unroll
    for (int ni = 0; ni < 4; ++ni) {
      const bf16_t* pB = &Bt[p][(wn * 64 + ni * 16 + nl) * LDB + half * 8];
      bfr[ni] = mk16(*(const bf16x8*)pB, *(const bf16x8*)(pB + 16));
    }
#pragma unroll
    for (int mi = 0; mi < 2; ++mi)
#pragma unroll
      for (int ni = 0; ni < 4; ++ni)
        acc[mi][ni] = wmma_bf16(af[mi], bfr[ni], acc[mi][ni]);

    if (more) {  // scatter-transpose next B tile while WMMAs drain
#pragma unroll
      for (int j = 0; j < 4; ++j) {
        int c = tid + j * 256, kk = c >> 5, n8 = (c & 31) * 8;
#pragma unroll
        for (int e = 0; e < 8; ++e) Bt[p ^ 1][(n8 + e) * LDB + kk] = breg[j][e];
      }
    }
    wait_async0();
    __syncthreads();
  }

  // ---- epilogue. C/D layout: vgpr v -> row (v + 8*(lane>>4)), col = lane&15.
#pragma unroll
  for (int mi = 0; mi < 2; ++mi)
#pragma unroll
    for (int ni = 0; ni < 4; ++ni)
#pragma unroll
      for (int v = 0; v < 8; ++v) {
        const int row = m0 + wm * 32 + mi * 16 + half * 8 + v;
        const int col = n0 + wn * 64 + ni * 16 + nl;
        const float val = acc[mi][ni][v];
        if (EPI == EPI_QKV) {
          const int s = col / DIM, rem = col - s * DIM;
          const int h = rem >> 6, d = rem & 63;
          const int b = row >> 10, n = row & (SEQ - 1);
          const size_t bh = (size_t)(b * HEADS + h);
          const bf16_t bv = f2bf(val);
          if (s == 0)      qb [(bh * SEQ + n) * 64 + d] = bv;
          else if (s == 1) kb [(bh * SEQ + n) * 64 + d] = bv;
          else             vtb[(bh * 64 + d) * SEQ + n] = bv;   // V transposed
        } else if (EPI == EPI_BIAS_RES) {
          const size_t idx = (size_t)row * N + col;
          outF[idx] = val + bias[col] + resid[idx];
        } else {  // EPI_GELU_BF (exact gelu)
          const float t = val + bias[col];
          outBf[(size_t)row * N + col] =
              f2bf(0.5f * t * (1.0f + erff(t * 0.70710678118654752f)));
        }
      }
}

// ---------------------------------------------------------------------------
// Attention: one block per (b, h, 32-query slab). 32x1024 fp32 score panel +
// 32x1024 bf16 P panel + Q tile resident in ~205KB of the 320KB WGP LDS.
// Q tile fetched by the Tensor Data Mover (TENSORcnt) when available.
// ---------------------------------------------------------------------------
#define LDS_S (SEQ + 8)   /* 1032 */
#define LDQ   72          /* 64 + 4-DWORD TDM pad per 32-DWORD row */
#define SMEM_Q_OFF ((size_t)32 * LDS_S * 6 + 2 * 32 * 8 * sizeof(float))
#define SMEM_TOTAL (SMEM_Q_OFF + (size_t)32 * LDQ * 2)

__global__ __launch_bounds__(256)
void attn_wmma(const bf16_t* __restrict__ q, const bf16_t* __restrict__ k,
               const bf16_t* __restrict__ vT, bf16_t* __restrict__ o) {
  extern __shared__ __align__(16) unsigned char smem[];
  float*  S    = (float*)smem;                                     // [32][1032] f32
  bf16_t* P    = (bf16_t*)(smem + (size_t)32 * LDS_S * 4);         // [32][1032] bf16
  float*  redA = (float*)(smem + (size_t)32 * LDS_S * 6);          // [32][8]
  float*  redB = redA + 32 * 8;
  bf16_t* Qs   = (bf16_t*)(smem + SMEM_Q_OFF);                     // [32][72]

  const int tid  = threadIdx.x;
  const int wid  = tid >> 5, lane = tid & 31;
  const int half = lane >> 4, nl  = lane & 15;
  const int b = blockIdx.z, h = blockIdx.y;
  const int q0 = blockIdx.x * 32;
  const size_t bh = (size_t)(b * HEADS + h);
  const bf16_t* qh = q  + bh * SEQ * 64;
  const bf16_t* kh = k  + bh * SEQ * 64;
  const bf16_t* vh = vT + bh * 64 * SEQ;

  // ---- Stage Q tile (32x64 bf16) into LDS via TDM
#if HAVE_TDM
  if (wid == 0) {
    // tile 64(x) x 32(y), row stride 64, pad 4 DWORDs every 32 DWORDs (->LDQ)
    tdm_load_2d(qh + (size_t)q0 * 64, Qs, 64u, (unsigned)(SEQ - q0), 64u,
                64u, 32u, /*padInt 2^(4+1)=32dw*/ 4u, /*padAmt 4dw*/ 3u, 1u);
    __builtin_amdgcn_s_wait_tensorcnt(0);
  }
#else
  for (int i = tid; i < 32 * 64; i += 256) {
    int r = i >> 6, c = i & 63;
    Qs[r * LDQ + c] = qh[(size_t)(q0 + r) * 64 + c];
  }
#endif
  __syncthreads();

  // ---- Phase 1: S = (1/sqrt(64)) * Q K^T  (128 16x16 tiles over 8 waves)
  for (int t = wid; t < 2 * 64; t += 8) {
    const int qi = t >> 6, kj = t & 63;
    const bf16_t* pq = Qs + (qi * 16 + nl) * LDQ + half * 8;
    bf16x16 a0 = mk16(*(const bf16x8*)pq,        *(const bf16x8*)(pq + 16));
    bf16x16 a1 = mk16(*(const bf16x8*)(pq + 32), *(const bf16x8*)(pq + 48));
    const bf16_t* pk = kh + (size_t)(kj * 16 + nl) * 64 + half * 8;
    bf16x16 b0 = mk16(*(const bf16x8*)pk,        *(const bf16x8*)(pk + 16));
    bf16x16 b1 = mk16(*(const bf16x8*)(pk + 32), *(const bf16x8*)(pk + 48));
    fx8 s = (fx8)0.0f;
    s = wmma_bf16(a0, b0, s);
    s = wmma_bf16(a1, b1, s);
#pragma unroll
    for (int v = 0; v < 8; ++v)
      S[(qi * 16 + half * 8 + v) * LDS_S + kj * 16 + nl] = s[v] * 0.125f;
  }
  __syncthreads();

  // ---- Phase 2: row softmax (8 threads per row), P = bf16(softmax(S))
  {
    const int r = tid >> 3, c0 = tid & 7;
    float* Sr = S + r * LDS_S;
    float lm = -1e30f;
    for (int c = c0; c < SEQ; c += 8) lm = fmaxf(lm, Sr[c]);
    redA[r * 8 + c0] = lm;
    __syncthreads();
    float rm = -1e30f;
#pragma unroll
    for (int i = 0; i < 8; ++i) rm = fmaxf(rm, redA[r * 8 + i]);
    float ls = 0.f;
    for (int c = c0; c < SEQ; c += 8) ls += expf(Sr[c] - rm);
    redB[r * 8 + c0] = ls;
    __syncthreads();
    float rs = 0.f;
#pragma unroll
    for (int i = 0; i < 8; ++i) rs += redB[r * 8 + i];
    const float inv = 1.0f / rs;
    bf16_t* Pr = P + r * LDS_S;
    for (int c = c0; c < SEQ; c += 8) Pr[c] = f2bf(expf(Sr[c] - rm) * inv);
  }
  __syncthreads();

  // ---- Phase 3: O = P @ V   (one 16x16 tile per wave, K=1024 -> 32 WMMAs)
  {
    const int qi = wid & 1, dj = wid >> 1;
    fx8 acc = (fx8)0.0f;
    for (int kt = 0; kt < SEQ / 32; ++kt) {
      const bf16_t* pp = P + (qi * 16 + nl) * LDS_S + kt * 32 + half * 8;
      bf16x16 a = mk16(*(const bf16x8*)pp, *(const bf16x8*)(pp + 16));
      const bf16_t* pv = vh + (size_t)(dj * 16 + nl) * SEQ + kt * 32 + half * 8;
      bf16x16 bb = mk16(*(const bf16x8*)pv, *(const bf16x8*)(pv + 16));
      acc = wmma_bf16(a, bb, acc);
    }
#pragma unroll
    for (int v = 0; v < 8; ++v) {
      const int qrow = q0 + qi * 16 + half * 8 + v;
      const int col  = h * 64 + dj * 16 + nl;
      o[((size_t)b * SEQ + qrow) * DIM + col] = f2bf(acc[v]);
    }
  }
}

// ---------------------------------------------------------------------------
extern "C" void kernel_launch(void* const* d_in, const int* in_sizes, int n_in,
                              void* d_out, int out_size, void* d_ws, size_t ws_size,
                              hipStream_t stream) {
  const float* x      = (const float*)d_in[0];
  const float* ln1_g  = (const float*)d_in[1];
  const float* ln1_b  = (const float*)d_in[2];
  const float* qkv_w  = (const float*)d_in[3];
  const float* proj_w = (const float*)d_in[4];
  const float* proj_b = (const float*)d_in[5];
  const float* ln2_g  = (const float*)d_in[6];
  const float* ln2_b  = (const float*)d_in[7];
  const float* fc1_w  = (const float*)d_in[8];
  const float* fc1_b  = (const float*)d_in[9];
  const float* fc2_w  = (const float*)d_in[10];
  const float* fc2_b  = (const float*)d_in[11];
  float* out = (float*)d_out;
  (void)in_sizes; (void)n_in; (void)out_size; (void)ws_size;

  unsigned char* ws = (unsigned char*)d_ws;
  size_t off = 0;
  auto alloc = [&](size_t bytes) -> void* {
    void* p = ws + off;
    off += (bytes + 255) & ~(size_t)255;
    return p;
  };
  bf16_t* wq    = (bf16_t*)alloc((size_t)DIM * 3 * DIM * 2);
  bf16_t* wproj = (bf16_t*)alloc((size_t)DIM * DIM * 2);
  bf16_t* wfc1  = (bf16_t*)alloc((size_t)DIM * HIDDEN * 2);
  bf16_t* wfc2  = (bf16_t*)alloc((size_t)HIDDEN * DIM * 2);
  bf16_t* h1    = (bf16_t*)alloc((size_t)ROWS * DIM * 2);
  bf16_t* qb    = (bf16_t*)alloc((size_t)BATCH * HEADS * SEQ * 64 * 2);
  bf16_t* kb    = (bf16_t*)alloc((size_t)BATCH * HEADS * SEQ * 64 * 2);
  bf16_t* vtb   = (bf16_t*)alloc((size_t)BATCH * HEADS * 64 * SEQ * 2);
  bf16_t* ob    = (bf16_t*)alloc((size_t)ROWS * DIM * 2);
  float*  x2    = (float*) alloc((size_t)ROWS * DIM * 4);
  bf16_t* h2    = (bf16_t*)alloc((size_t)ROWS * DIM * 2);
  bf16_t* h3    = (bf16_t*)alloc((size_t)ROWS * HIDDEN * 2);

  {
    int n;
    n = DIM * 3 * DIM;   cvt_bf16<<<(n + 255) / 256, 256, 0, stream>>>(qkv_w,  wq,    n);
    n = DIM * DIM;       cvt_bf16<<<(n + 255) / 256, 256, 0, stream>>>(proj_w, wproj, n);
    n = DIM * HIDDEN;    cvt_bf16<<<(n + 255) / 256, 256, 0, stream>>>(fc1_w,  wfc1,  n);
    n = HIDDEN * DIM;    cvt_bf16<<<(n + 255) / 256, 256, 0, stream>>>(fc2_w,  wfc2,  n);
  }

  ln_bf16<<<ROWS, 256, 0, stream>>>(x, ln1_g, ln1_b, h1);

  gemm_bf16_wmma<EPI_QKV><<<dim3((3 * DIM) / BN, ROWS / BM), 256, 0, stream>>>(
      h1, wq, ROWS, 3 * DIM, DIM, nullptr, nullptr, nullptr, nullptr, qb, kb, vtb);

  attn_wmma<<<dim3(SEQ / 32, HEADS, BATCH), 256, SMEM_TOTAL, stream>>>(qb, kb, vtb, ob);

  gemm_bf16_wmma<EPI_BIAS_RES><<<dim3(DIM / BN, ROWS / BM), 256, 0, stream>>>(
      ob, wproj, ROWS, DIM, DIM, proj_b, x, x2, nullptr, nullptr, nullptr, nullptr);

  ln_bf16<<<ROWS, 256, 0, stream>>>(x2, ln2_g, ln2_b, h2);

  gemm_bf16_wmma<EPI_GELU_BF><<<dim3(HIDDEN / BN, ROWS / BM), 256, 0, stream>>>(
      h2, wfc1, ROWS, HIDDEN, DIM, fc1_b, nullptr, nullptr, h3, nullptr, nullptr, nullptr);

  gemm_bf16_wmma<EPI_BIAS_RES><<<dim3(DIM / BN, ROWS / BM), 256, 0, stream>>>(
      h3, wfc2, ROWS, DIM, HIDDEN, fc2_b, x2, out, nullptr, nullptr, nullptr, nullptr);
}
